// SwitchHeadAttention_4045859193472
// MI455X (gfx1250) — matrix operand order repaired and checked by reading, not compiled, not run on hardware
//
#include <hip/hip_runtime.h>
#include <math.h>

// ---------------------------------------------------------------------------
// Problem constants
// ---------------------------------------------------------------------------
#define B_    2
#define T_    2048
#define DIM_  1024
#define H_    8
#define E_    5
#define D_    64
#define K_    3
#define ROWS  (B_ * T_)          // 4096
#define NQ    (H_ * E_ * D_)     // 2560
#define NKV   (2 * H_ * E_ * D_) // 5120
#define HD    (H_ * D_)          // 512  (attn-out reduction dim)

typedef __bf16 bf16;
typedef __attribute__((ext_vector_type(8)))  bf16  v8bf;
typedef __attribute__((ext_vector_type(16))) bf16  v16bf;
typedef __attribute__((ext_vector_type(8)))  float v8f;

static __device__ __forceinline__ unsigned bf_bits(float f) {
    bf16 b = (bf16)f;
    unsigned short u;
    __builtin_memcpy(&u, &b, 2);
    return (unsigned)u;
}

static __device__ __forceinline__ v16bf cat8(v8bf lo, v8bf hp) {
    return __builtin_shufflevector(lo, hp, 0, 1, 2, 3, 4, 5, 6, 7,
                                   8, 9, 10, 11, 12, 13, 14, 15);
}

// ---------------------------------------------------------------------------
// Workspace layout (bytes, all 256-aligned)
// ---------------------------------------------------------------------------
#define OFF_XB    ((size_t)0)          //  x  bf16 [4096][1024]
#define OFF_WQB   ((size_t)8388608)    //  w_q^T  bf16 [2560][1024]
#define OFF_WKVB  ((size_t)13631488)   //  w_kv^T bf16 [5120][1024]
#define OFF_WOB   ((size_t)24117248)   //  w_o^T  bf16 [1024][512]
#define OFF_BIAS  ((size_t)25165824)   //  bias f32 [1024]
#define OFF_WTS   ((size_t)25169920)   //  wt_s f32 [4096][8][5]
#define OFF_WTD   ((size_t)25825280)   //  wt_d f32 [4096][8][5]
#define OFF_QF    ((size_t)26480640)   //  q_full  bf16 [4096][2560]
#define OFF_KVF   ((size_t)47452160)   //  kv_full bf16 [4096][5120]
#define OFF_QH    ((size_t)89395200)   //  q bf16 [B*H][T][D]   (scale folded)
#define OFF_KH    ((size_t)93589504)   //  k bf16 [B*H][T][D]
#define OFF_VT    ((size_t)97783808)   //  v bf16 [B*H][D][T]   (transposed)
#define OFF_AO    ((size_t)101978112)  //  attn out bf16 [4096][512]

// ---------------------------------------------------------------------------
// Elementwise conversions / packing
// ---------------------------------------------------------------------------
__global__ void k_f32_to_bf16(const float* __restrict__ in, bf16* __restrict__ out, int n) {
    int i = blockIdx.x * blockDim.x + threadIdx.x;
    if (i < n) out[i] = (bf16)in[i];
}

// in: (K x N) row-major f32  ->  out: (N x K) row-major bf16
__global__ void k_transpose_bf16(const float* __restrict__ in, bf16* __restrict__ out,
                                 int K, int N) {
    int i = blockIdx.x * blockDim.x + threadIdx.x;
    if (i >= K * N) return;
    int k = i / N, n = i - k * N;
    out[(size_t)n * K + k] = (bf16)in[i];
}

// w_o_w (H,DIM,D) -> wob[f][h*64+d] bf16
__global__ void k_pack_wo(const float* __restrict__ wow, bf16* __restrict__ out) {
    int i = blockIdx.x * blockDim.x + threadIdx.x;
    if (i >= DIM_ * HD) return;
    int f = i / HD, hd = i - f * HD;
    int h = hd >> 6, d = hd & 63;
    out[i] = (bf16)wow[((size_t)h * DIM_ + f) * D_ + d];
}

__global__ void k_bias(const float* __restrict__ wob_b, float* __restrict__ bias) {
    int f = blockIdx.x * blockDim.x + threadIdx.x;
    if (f >= DIM_) return;
    float s = 0.f;
#pragma unroll
    for (int h = 0; h < H_; ++h) s += wob_b[h * DIM_ + f];
    bias[f] = s;
}

// ---------------------------------------------------------------------------
// Gating: sigmoid(x@w_s), sigmoid(x@w_d), top-3-of-5, dense weight tables.
// Weight for expert eps[k] is the raw score of expert *position* k (scatter
// semantics of the reference). One block (128 thr) per row.
// ---------------------------------------------------------------------------
__global__ void k_gating(const float* __restrict__ x, const float* __restrict__ w_s,
                         const float* __restrict__ w_d,
                         float* __restrict__ wts, float* __restrict__ wtd) {
    __shared__ float xs[DIM_];
    __shared__ float sc[2 * H_ * E_]; // [gate][h*5+e], gate0=s, gate1=d
    const int tid = threadIdx.x;
    const int row = blockIdx.x;
    for (int i = tid; i < DIM_; i += 128) xs[i] = x[(size_t)row * DIM_ + i];
    __syncthreads();
    if (tid < 80) {
        const float* w = (tid < 40) ? w_s : w_d;
        const int c = (tid < 40) ? tid : tid - 40;
        float acc = 0.f;
#pragma unroll 8
        for (int k = 0; k < DIM_; ++k) acc += xs[k] * w[k * 40 + c];
        sc[tid] = 1.f / (1.f + __expf(-acc));
    }
    __syncthreads();
    if (tid < 16) {
        const int g = tid >> 3, h = tid & 7;
        float s[E_];
#pragma unroll
        for (int e = 0; e < E_; ++e) s[e] = sc[g * 40 + h * E_ + e];
        float wt[E_] = {0.f, 0.f, 0.f, 0.f, 0.f};
        bool used[E_] = {false, false, false, false, false};
#pragma unroll
        for (int k = 0; k < K_; ++k) {  // argmax; strict '>' keeps lowest index on ties
            int best = 0; float bv = -1e30f;
#pragma unroll
            for (int e = 0; e < E_; ++e)
                if (!used[e] && s[e] > bv) { bv = s[e]; best = e; }
            used[best] = true;
            wt[best] = s[k];            // raw score of position k (reference quirk)
        }
        float* dst = ((g == 0) ? wts : wtd) + ((size_t)row * H_ + h) * E_;
#pragma unroll
        for (int e = 0; e < E_; ++e) dst[e] = wt[e];
    }
}

// ---------------------------------------------------------------------------
// Generic bf16 WMMA GEMM:  C(MxN) = A(MxK) * Bt(NxK)^T  [+ bias]
// Block = 128 threads (4 waves); wave tile = 32x64; block tile = 32x256.
// Requires K % 32 == 0, N % 256 == 0, M % 32 == 0.
// ---------------------------------------------------------------------------
template <bool OUTF32, bool HASBIAS>
__global__ void k_gemm_wmma(const bf16* __restrict__ A, const bf16* __restrict__ Bt,
                            void* __restrict__ Cout, const float* __restrict__ bias,
                            int M, int N, int Kd) {
    const int lane = threadIdx.x & 31;
    const int wave = threadIdx.x >> 5;
    const int r = lane & 15, hi = lane >> 4;
    const int nb = blockIdx.x * 256 + wave * 64;
    const int row0 = blockIdx.y * 32;

    v8f acc[2][4];
#pragma unroll
    for (int mt = 0; mt < 2; ++mt)
#pragma unroll
        for (int t = 0; t < 4; ++t) acc[mt][t] = v8f{};

    const bf16* ap0 = A + (size_t)(row0 + r) * Kd + hi * 8;
    const bf16* ap1 = ap0 + (size_t)16 * Kd;
    const bf16* bp  = Bt + (size_t)(nb + r) * Kd + hi * 8;
    const size_t bstride = (size_t)16 * Kd;

    for (int k0 = 0; k0 < Kd; k0 += 32) {
        v16bf a0 = cat8(*(const v8bf*)(ap0 + k0), *(const v8bf*)(ap0 + k0 + 16));
        v16bf a1 = cat8(*(const v8bf*)(ap1 + k0), *(const v8bf*)(ap1 + k0 + 16));
#pragma unroll
        for (int t = 0; t < 4; ++t) {
            v16bf b = cat8(*(const v8bf*)(bp + (size_t)t * bstride + k0), *(const v8bf*)(bp + (size_t)t * bstride + k0 + 16));
            acc[0][t] = __builtin_amdgcn_wmma_f32_16x16x32_bf16(
                false, a0, false, b, (short)0, acc[0][t], false, false);
            acc[1][t] = __builtin_amdgcn_wmma_f32_16x16x32_bf16(
                false, a1, false, b, (short)0, acc[1][t], false, false);
        }
    }

#pragma unroll
    for (int t = 0; t < 4; ++t) {
        const int col = nb + t * 16 + r;
        const float bs = HASBIAS ? bias[col] : 0.f;
#pragma unroll
        for (int mt = 0; mt < 2; ++mt) {
#pragma unroll
            for (int v = 0; v < 8; ++v) {
                const int row = row0 + mt * 16 + hi * 8 + v;
                const float val = acc[mt][t][v] + bs;
                if (OUTF32)
                    ((float*)Cout)[(size_t)row * N + col] = val;
                else
                    ((bf16*)Cout)[(size_t)row * N + col] = (bf16)val;
            }
        }
    }
}

// ---------------------------------------------------------------------------
// Expert select: q/k/v = sum_e wt[e] * proj[...,e,:]  (wt dense, 0 unselected)
// q gets the 1/sqrt(D) scale folded in.  One thread per (b,t,h,d).
// ---------------------------------------------------------------------------
__global__ void k_select(const bf16* __restrict__ qf, const bf16* __restrict__ kvf,
                         const float* __restrict__ wts, const float* __restrict__ wtd,
                         bf16* __restrict__ qh, bf16* __restrict__ kh,
                         bf16* __restrict__ vt) {
    const int idx = blockIdx.x * blockDim.x + threadIdx.x;
    if (idx >= ROWS * H_ * D_) return;
    const int d = idx & 63;
    const int h = (idx >> 6) & 7;
    const int t = (idx >> 9) & 2047;
    const int b = (idx >> 20) & 1;
    const int row = b * T_ + t;

    const float* wd = wtd + ((size_t)row * H_ + h) * E_;
    const float* ws = wts + ((size_t)row * H_ + h) * E_;
    const size_t bq = (size_t)row * NQ + h * (E_ * D_) + d;
    const size_t bk = (size_t)row * NKV + h * (E_ * D_) + d;
    const size_t bv = bk + (size_t)H_ * E_ * D_;

    float q = 0.f, k = 0.f, v = 0.f;
#pragma unroll
    for (int e = 0; e < E_; ++e) {
        q += wd[e] * (float)qf[bq + e * D_];
        k += ws[e] * (float)kvf[bk + e * D_];
        v += ws[e] * (float)kvf[bv + e * D_];
    }
    const int bh = b * H_ + h;
    qh[((size_t)bh * T_ + t) * D_ + d] = (bf16)(q * 0.125f); // fold D^-0.5
    kh[((size_t)bh * T_ + t) * D_ + d] = (bf16)k;
    vt[((size_t)bh * D_ + d) * T_ + t] = (bf16)v;
}

// ---------------------------------------------------------------------------
// Flash attention, transposed formulation:
//   S^T = K * Q^T      (query lives in the lane column -> softmax is in-lane)
//   out^T = V^T * P^T  (P^T B-frag assembled in registers via 8 packed shfls)
// One wave per 16-query i-tile; j-step = 32; no LDS.
// ---------------------------------------------------------------------------
__global__ void k_attn(const bf16* __restrict__ qh, const bf16* __restrict__ kh,
                       const bf16* __restrict__ vt, bf16* __restrict__ ao) {
    const int lane = threadIdx.x & 31;
    const int wave = threadIdx.x >> 5;
    const int r = lane & 15, hi = lane >> 4;
    const int tile = blockIdx.x * 4 + wave;
    const int bh = tile >> 7;            // T/16 = 128 i-tiles per (b,h)
    const int i0 = (tile & 127) * 16;

    const bf16* qb = qh + (size_t)bh * T_ * D_;
    const bf16* kb = kh + (size_t)bh * T_ * D_;
    const bf16* vb = vt + (size_t)bh * D_ * T_;

    // Q^T B-frags, held for the whole loop: lane = query column i0+r,
    // K-run = d in [c*32 + hi*8, +8) and [c*32 + 16 + hi*8, +8)
    v16bf bq[2];
#pragma unroll
    for (int c = 0; c < 2; ++c)
        bq[c] = cat8(*(const v8bf*)(qb + (size_t)(i0 + r) * D_ + c * 32 + hi * 8), *(const v8bf*)(qb + (size_t)(i0 + r) * D_ + c * 32 + hi * 8 + 16));

    float m = -INFINITY;
    float l = 0.f;                        // per-half partial; combined at end
    v8f acc[4] = {v8f{}, v8f{}, v8f{}, v8f{}};

    for (int j0 = 0; j0 < T_; j0 += 32) {
        // S^T tiles: st[jt] covers keys j0+jt*16 .. +15 (rows), queries (cols)
        v8f st[2] = {v8f{}, v8f{}};
#pragma unroll
        for (int jt = 0; jt < 2; ++jt)
#pragma unroll
            for (int c = 0; c < 2; ++c) {
                const bf16* p = kb + (size_t)(j0 + jt * 16 + r) * D_ + c * 32 + hi * 8;
                v16bf ak = cat8(*(const v8bf*)p, *(const v8bf*)(p + 16));
                st[jt] = __builtin_amdgcn_wmma_f32_16x16x32_bf16(
                    false, ak, false, bq[c], (short)0, st[jt], false, false);
            }

        // ---- online softmax: all 16 s-values for this query are in-lane ----
        float mx = st[0][0];
#pragma unroll
        for (int v = 1; v < 8; ++v) mx = fmaxf(mx, st[0][v]);
#pragma unroll
        for (int v = 0; v < 8; ++v) mx = fmaxf(mx, st[1][v]);
        mx = fmaxf(mx, __shfl_xor(mx, 16, 32));   // other half's 16 j-values
        const float mnew = fmaxf(m, mx);
        const float al = __expf(m - mnew);
        m = mnew;

        float p0[8], p1[8], rs = 0.f;
#pragma unroll
        for (int v = 0; v < 8; ++v) {
            p0[v] = __expf(st[0][v] - mnew);
            p1[v] = __expf(st[1][v] - mnew);
            rs += p0[v] + p1[v];
        }
        l = l * al + rs;                  // half-local sum (linear in alpha)

#pragma unroll
        for (int t = 0; t < 4; ++t)
#pragma unroll
            for (int v = 0; v < 8; ++v) acc[t][v] *= al;  // al uniform per query

        // ---- assemble P^T B-frag: lane needs j = hi*16 .. hi*16+15 --------
        // this half owns j = hi*8+v (p0) and 16+hi*8+v (p1); other 8+8 come
        // from lane^16.  Exchange packed bf16 pairs: 8 bpermutes, batched.
        unsigned pk0[4], pk1[4], ex0[4], ex1[4];
#pragma unroll
        for (int j = 0; j < 4; ++j) {
            pk0[j] = bf_bits(p0[2 * j]) | (bf_bits(p0[2 * j + 1]) << 16);
            pk1[j] = bf_bits(p1[2 * j]) | (bf_bits(p1[2 * j + 1]) << 16);
        }
#pragma unroll
        for (int j = 0; j < 4; ++j) ex0[j] = __shfl_xor(pk0[j], 16, 32);
#pragma unroll
        for (int j = 0; j < 4; ++j) ex1[j] = __shfl_xor(pk1[j], 16, 32);

        union { v16bf v; unsigned u[8]; } Bp;
#pragma unroll
        for (int j = 0; j < 4; ++j) {
            Bp.u[j]     = hi ? ex1[j] : pk0[j];  // j-run first  8 of this half
            Bp.u[4 + j] = hi ? pk1[j] : ex0[j];  // j-run second 8
        }

        // ---- out^T += V^T * P^T  (4 d-tiles of 16 rows) --------------------
#pragma unroll
        for (int td = 0; td < 4; ++td) {
            const bf16* pA = vb + (size_t)(td * 16 + r) * T_ + j0 + hi * 16;
            v16bf av = *(const v16bf*)pA;
            acc[td] = __builtin_amdgcn_wmma_f32_16x16x32_bf16(
                false, av, false, Bp.v, (short)0, acc[td], false, false);
        }
    }

    // combine the two halves' partial sums, normalize, store (paired bf16)
    const float ltot = l + __shfl_xor(l, 16, 32);
    const float inv = 1.f / ltot;
    const int b = bh >> 3, h = bh & 7;
    const size_t rowbase = (size_t)(b * T_ + i0 + r) * HD + h * D_;
#pragma unroll
    for (int td = 0; td < 4; ++td)
#pragma unroll
        for (int v = 0; v < 8; v += 2) {
            unsigned pk = bf_bits(acc[td][v] * inv) |
                          (bf_bits(acc[td][v + 1] * inv) << 16);
            *(unsigned*)((char*)ao + 2 * (rowbase + td * 16 + hi * 8 + v)) = pk;
        }
}

// ---------------------------------------------------------------------------
// Launch
// ---------------------------------------------------------------------------
extern "C" void kernel_launch(void* const* d_in, const int* in_sizes, int n_in,
                              void* d_out, int out_size, void* d_ws, size_t ws_size,
                              hipStream_t stream) {
    const float* x     = (const float*)d_in[0];
    const float* w_q   = (const float*)d_in[1];
    const float* w_kv  = (const float*)d_in[2];
    const float* w_s   = (const float*)d_in[3];
    const float* w_d   = (const float*)d_in[4];
    const float* wow   = (const float*)d_in[5];
    const float* wob_b = (const float*)d_in[6];
    float* out = (float*)d_out;
    char* ws = (char*)d_ws;

    bf16*  xb   = (bf16*)(ws + OFF_XB);
    bf16*  wqb  = (bf16*)(ws + OFF_WQB);
    bf16*  wkvb = (bf16*)(ws + OFF_WKVB);
    bf16*  wob  = (bf16*)(ws + OFF_WOB);
    float* bias = (float*)(ws + OFF_BIAS);
    float* wts  = (float*)(ws + OFF_WTS);
    float* wtd  = (float*)(ws + OFF_WTD);
    bf16*  qf   = (bf16*)(ws + OFF_QF);
    bf16*  kvf  = (bf16*)(ws + OFF_KVF);
    bf16*  qh   = (bf16*)(ws + OFF_QH);
    bf16*  kh   = (bf16*)(ws + OFF_KH);
    bf16*  vt   = (bf16*)(ws + OFF_VT);
    bf16*  ao   = (bf16*)(ws + OFF_AO);

    // 1. conversions / packing
    k_f32_to_bf16<<<(ROWS * DIM_ + 255) / 256, 256, 0, stream>>>(x, xb, ROWS * DIM_);
    k_transpose_bf16<<<(DIM_ * NQ + 255) / 256, 256, 0, stream>>>(w_q, wqb, DIM_, NQ);
    k_transpose_bf16<<<(DIM_ * NKV + 255) / 256, 256, 0, stream>>>(w_kv, wkvb, DIM_, NKV);
    k_pack_wo<<<(DIM_ * HD + 255) / 256, 256, 0, stream>>>(wow, wob);
    k_bias<<<(DIM_ + 255) / 256, 256, 0, stream>>>(wob_b, bias);

    // 2. gating + weight tables
    k_gating<<<ROWS, 128, 0, stream>>>(x, w_s, w_d, wts, wtd);

    // 3. projections (WMMA)
    k_gemm_wmma<false, false><<<dim3(NQ / 256, ROWS / 32), 128, 0, stream>>>(
        xb, wqb, (void*)qf, nullptr, ROWS, NQ, DIM_);
    k_gemm_wmma<false, false><<<dim3(NKV / 256, ROWS / 32), 128, 0, stream>>>(
        xb, wkvb, (void*)kvf, nullptr, ROWS, NKV, DIM_);

    // 4. expert select -> q,k,v
    k_select<<<(ROWS * H_ * D_ + 255) / 256, 256, 0, stream>>>(qf, kvf, wts, wtd,
                                                               qh, kh, vt);

    // 5. flash attention (WMMA, transposed, no LDS)
    k_attn<<<(B_ * H_ * (T_ / 16)) / 4, 128, 0, stream>>>(qh, kh, vt, ao);

    // 6. output projection + bias (WMMA, f32 out)
    k_gemm_wmma<true, true><<<dim3(DIM_ / 256, ROWS / 32), 128, 0, stream>>>(
        ao, wob, (void*)out, bias, ROWS, DIM_, HD);
}
